// GraphUnet_39436389712231
// MI455X (gfx1250) — matrix operand order repaired and checked by reading, not compiled, not run on hardware
//
#include <hip/hip_runtime.h>
#include <math.h>

typedef __attribute__((ext_vector_type(16))) _Float16 v16h;
typedef __attribute__((ext_vector_type(8)))  _Float16 v8h;
typedef __attribute__((ext_vector_type(8)))  float    v8f;

#define NODES 4096
#define DIM   320
#define NL1   3276   // int(0.8*4096)
#define NL2   1965   // int(0.6*3276)
#define WSLOT (DIM * DIM)   // one 320x320 swizzled weight slot (halves)

__device__ __forceinline__ float actf(float v, int act) {
  if (act == 1) return v > 0.f ? v : 0.f;
  if (act == 2) return 1.f / (1.f + __expf(-v));
  return v;
}

// f32 -> f16 row-major conversion (activations, once per GEMM)
__global__ void cvt_f16(const float* __restrict__ src, _Float16* __restrict__ dst, int n)
{
  int t = blockIdx.x * blockDim.x + threadIdx.x;
  if (t < n) dst[t] = (_Float16)src[t];
}

// f32 weight (KxN, N=320, K mult of 32) -> f16 swizzled into WMMA B-fragment
// order: Bsw[((kb*20 + T)*32 + lane)*16 + p] = B[kb*32 + (lane>=16?16:0) + p][T*16 + (lane&15)]
// so a GEMM wave reads each B fragment as one contiguous 32B v16h per lane.
__global__ void swizzleB(const float* __restrict__ B, _Float16* __restrict__ Bsw, int K)
{
  int t = blockIdx.x * blockDim.x + threadIdx.x;
  if (t >= K * DIM) return;
  int p    = t & 15;
  int lane = (t >> 4) & 31;
  int T    = (t >> 9) % 20;
  int kb   = t / (512 * 20);
  int k    = kb * 32 + ((lane >> 4) << 4) + p;
  int n    = T * 16 + (lane & 15);
  Bsw[t] = (_Float16)B[(size_t)k * DIM + n];
}

// ---------------------------------------------------------------------------
// Fused GEMM on v_wmma_f32_16x16x32_f16 (wave32). N = 320 fixed.
//   out = act( A[MxK]@B + (A2[MxK2]@B2) + bias + preadd ) + postadd
// Block = 256 thr = 8 waves -> 32 rows x 320 cols. Wave w: mtile=w&1,
// colgroup=w>>1 (5 WMMA tiles, A fragment reused across all 5).
// A k-slab (32x32 f16 = 2KB) staged in LDS with global_load_async_to_lds_b128;
// A fragment = 2x ds_load_b128. B pre-swizzled -> fragment = 2x global b128.
// ---------------------------------------------------------------------------
__global__ void gemm_wmma(const _Float16* __restrict__ Ah, const _Float16* __restrict__ Bsw,
                          const _Float16* __restrict__ A2h, const _Float16* __restrict__ B2sw,
                          const float* __restrict__ bias,
                          const float* __restrict__ preadd,
                          const float* __restrict__ postadd,
                          float* __restrict__ out,
                          int M, int N, int K, int K2, int act)
{
  __shared__ __align__(16) _Float16 sA[32 * 32];   // 2KB A slab: [row][32 k]

  const int tid    = threadIdx.x;
  const int lane   = tid & 31;
  const int wave   = tid >> 5;
  const int half16 = lane >> 4;
  const int l15    = lane & 15;
  const int m0     = blockIdx.x * 32 + (wave & 1) * 16;
  const int cg     = wave >> 1;             // column group: 5 tiles of 16 = 80 cols
  const int n0     = cg * 80;
  const int mrow   = (wave & 1) * 16 + l15; // local A-slab row for this lane

  v8f acc[5] = {};
  if (preadd) {
#pragma unroll
    for (int t = 0; t < 5; ++t)
#pragma unroll
      for (int r = 0; r < 8; ++r) {
        int m = m0 + r + half16 * 8;
        if (m < M) acc[t][r] = preadd[(size_t)m * N + n0 + t * 16 + l15];
      }
  }

  const unsigned ldsA = (unsigned)(size_t)(&sA[0]);

  for (int pass = 0; pass < 2; ++pass) {
    const _Float16* Ap = pass ? A2h  : Ah;
    const _Float16* Bp = pass ? B2sw : Bsw;
    const int       Kp = pass ? K2   : K;
    if (!Ap) continue;                       // kernel-arg uniform

    for (int k0 = 0; k0 < Kp; k0 += 32) {
      __syncthreads();                       // previous slab fully consumed
      if (tid < 128) {                       // 32 rows x 4 b128 chunks
        int r = tid >> 2, piece = tid & 3;
        int gr = blockIdx.x * 32 + r; if (gr > M - 1) gr = M - 1;
        const char* g    = (const char*)(Ap + (size_t)gr * Kp + k0 + piece * 8);
        unsigned    ldst = ldsA + (unsigned)(r * 64 + piece * 16);
        asm volatile("global_load_async_to_lds_b128 %0, %1, off"
                     :: "v"(ldst), "v"(g) : "memory");
      }
      asm volatile("s_wait_asynccnt 0" ::: "memory");
      __syncthreads();                       // slab visible to all waves

      // A fragment: two contiguous 16B LDS reads, reused by all 5 WMMAs
      v8h  alo = *(const v8h*)(sA + mrow * 32 + half16 * 16);
      v8h  ahi = *(const v8h*)(sA + mrow * 32 + half16 * 16 + 8);
      v16h av  = __builtin_shufflevector(alo, ahi,
                     0,1,2,3,4,5,6,7,8,9,10,11,12,13,14,15);

      const _Float16* bk = Bp + (size_t)(k0 >> 5) * (20 * 512);
#pragma unroll
      for (int t = 0; t < 5; ++t) {
        int T = cg * 5 + t;
        v16h bv = *(const v16h*)(bk + ((size_t)T * 32 + lane) * 16);
        acc[t] = __builtin_amdgcn_wmma_f32_16x16x32_f16(false, av, false, bv,
                                                        (short)0, acc[t], false, false);
      }
    }
  }

#pragma unroll
  for (int t = 0; t < 5; ++t) {
    int nn = n0 + t * 16 + l15;
#pragma unroll
    for (int r = 0; r < 8; ++r) {
      int m = m0 + r + half16 * 8;
      if (m < M) {
        float v = acc[t][r];
        if (bias) v += bias[nn];
        v = actf(v, act);
        if (postadd) v += postadd[(size_t)m * N + nn];
        out[(size_t)m * N + nn] = v;
      }
    }
  }
}

// Hp[:,0] = X @ pW[lvl] + pwb[lvl] + pb[lvl]   (one thread per row)
__global__ void matvec_bias2(const float* __restrict__ X, const float* __restrict__ w,
                             const float* __restrict__ c1, const float* __restrict__ c2,
                             int lvl, float* __restrict__ h, int M, int K)
{
  int m = blockIdx.x * blockDim.x + threadIdx.x;
  if (m >= M) return;
  float s = c1[lvl] + c2[lvl];
  const float* x = X + (size_t)m * K;
  for (int k = 0; k < K; ++k) s += x[k] * w[k];
  h[m] = s;
}

// Masked row softmax + pooled score. NEG entries underflow to exactly 0 in f32,
// so only unmasked columns (A[r,c]!=0 || c==r) contribute.
__global__ void pool_scores(const float* __restrict__ A, const float* __restrict__ h,
                            const float* __restrict__ phi, int lvl, int n,
                            float* __restrict__ scores)
{
  __shared__ float red[256];
  const int r = blockIdx.x;
  const float p0 = phi[lvl * 2 + 0];
  const float p1 = phi[lvl * 2 + 1];
  const float hr = h[r];
  const float* arow = A + (size_t)r * n;

  float mx = -3.0e38f;
  for (int c = threadIdx.x; c < n; c += blockDim.x)
    if (arow[c] != 0.f || c == r)
      mx = fmaxf(mx, p0 * h[c] + p1 * hr);
  red[threadIdx.x] = mx; __syncthreads();
  for (int s = blockDim.x >> 1; s > 0; s >>= 1) {
    if (threadIdx.x < s) red[threadIdx.x] = fmaxf(red[threadIdx.x], red[threadIdx.x + s]);
    __syncthreads();
  }
  mx = red[0]; __syncthreads();

  float den = 0.f, num = 0.f;
  for (int c = threadIdx.x; c < n; c += blockDim.x)
    if (arow[c] != 0.f || c == r) {
      float e = __expf(p0 * h[c] + p1 * hr - mx);
      den += e; num += e * h[c];
    }
  red[threadIdx.x] = den; __syncthreads();
  for (int s = blockDim.x >> 1; s > 0; s >>= 1) {
    if (threadIdx.x < s) red[threadIdx.x] += red[threadIdx.x + s];
    __syncthreads();
  }
  den = red[0]; __syncthreads();
  red[threadIdx.x] = num; __syncthreads();
  for (int s = blockDim.x >> 1; s > 0; s >>= 1) {
    if (threadIdx.x < s) red[threadIdx.x] += red[threadIdx.x + s];
    __syncthreads();
  }
  num = red[0];
  if (threadIdx.x == 0)
    scores[r] = 1.f / (1.f + __expf(-(num / den) / 100.f));
}

// Exact jax.lax.top_k semantics via rank counting (descending, index ties).
__global__ void topk_rank(const float* __restrict__ s, int n, int kk,
                          int* __restrict__ idx, float* __restrict__ vals)
{
  int i = blockIdx.x * blockDim.x + threadIdx.x;
  if (i >= n) return;
  float si = s[i];
  int rank = 0;
  for (int j = 0; j < n; ++j) {
    float sj = s[j];
    rank += (sj > si) || (sj == si && j < i);
  }
  if (rank < kk) { idx[rank] = i; vals[rank] = si; }
}

__global__ void gather_rows_scale(const float* __restrict__ X, const int* __restrict__ idx,
                                  const float* __restrict__ vals, float* __restrict__ out,
                                  int m, int d)
{
  int t = blockIdx.x * blockDim.x + threadIdx.x;
  if (t >= m * d) return;
  int r = t / d, c = t - r * d;
  out[t] = X[(size_t)idx[r] * d + c] * vals[r];
}

__global__ void gather_adj(const float* __restrict__ A, const int* __restrict__ idx,
                           float* __restrict__ out, int m, int n_src)
{
  int t = blockIdx.x * blockDim.x + threadIdx.x;
  if (t >= m * m) return;
  int r = t / m, c = t - r * m;
  out[t] = A[(size_t)idx[r] * n_src + idx[c]];
}

__global__ void zero_buf(float* __restrict__ p, int n)
{
  for (int t = blockIdx.x * blockDim.x + threadIdx.x; t < n; t += gridDim.x * blockDim.x)
    p[t] = 0.f;
}

__global__ void scatter_rows(const float* __restrict__ X, const int* __restrict__ idx,
                             float* __restrict__ out, int m, int d)
{
  int t = blockIdx.x * blockDim.x + threadIdx.x;
  if (t >= m * d) return;
  int r = t / d, c = t - r * d;
  out[(size_t)idx[r] * d + c] = X[t];
}

extern "C" void kernel_launch(void* const* d_in, const int* in_sizes, int n_in,
                              void* d_out, int out_size, void* d_ws, size_t ws_size,
                              hipStream_t stream)
{
  const float* A_in = (const float*)d_in[0];
  const float* X_in = (const float*)d_in[1];
  const float* W0   = (const float*)d_in[2];
  const float* b0   = (const float*)d_in[3];
  const float* Wb   = (const float*)d_in[4];
  const float* bb   = (const float*)d_in[5];
  const float* We   = (const float*)d_in[6];
  const float* be   = (const float*)d_in[7];
  const float* dW   = (const float*)d_in[8];
  const float* db   = (const float*)d_in[9];
  const float* uW   = (const float*)d_in[10];
  const float* ub   = (const float*)d_in[11];
  const float* pW   = (const float*)d_in[12];
  const float* pwb  = (const float*)d_in[13];
  const float* pb   = (const float*)d_in[14];
  const float* phi  = (const float*)d_in[15];
  const float* agWg = (const float*)d_in[16];
  const float* agbg = (const float*)d_in[17];
  const float* agWs = (const float*)d_in[18];
  const float* agbs = (const float*)d_in[19];
  const float* agWp = (const float*)d_in[20];
  const float* agbp = (const float*)d_in[21];

  float* out   = (float*)d_out;                       // (4096,320)
  float* start = (float*)d_out + (size_t)NODES * DIM; // (4096,320), output #2

  // ---- workspace carve (256B aligned slots) ----
  float* ws = (float*)d_ws;
  size_t off = 0;
  auto take  = [&](size_t nf) { float* p = ws + off; off += (nf + 63) & ~(size_t)63; return p; };
  auto takeH = [&](size_t nh) { return (_Float16*)take((nh + 1) / 2); };
  const size_t FB = (size_t)NODES * DIM;
  float* down0 = take(FB);
  float* down1 = take(FB);
  float* Ap1   = take((size_t)NL1 * NL1);
  float* G0 = take(FB); float* G1 = take(FB); float* G2 = take(FB);
  float* G3 = take(FB); float* G4 = take(FB); float* G5 = take(FB);
  float* h      = take(NODES);
  float* scores = take(NODES);
  float* vals0  = take(NODES);
  float* vals1  = take(NODES);
  int* idx0 = (int*)take(NODES);
  int* idx1 = (int*)take(NODES);
  // swizzled f16 weight arena: 16 slots of 320x320
  _Float16* Bsw = takeH((size_t)16 * WSLOT);
  _Float16* AhA = takeH(FB);
  _Float16* AhB = takeH(FB);
  (void)ws_size; (void)in_sizes; (void)n_in; (void)out_size;

  auto g1  = [](int n) { return (n + 255) / 256; };
  auto cvt = [&](const float* s, _Float16* d, int n) {
    cvt_f16<<<g1(n), 256, 0, stream>>>(s, d, n);
  };
  auto slot = [&](int i) { return Bsw + (size_t)i * WSLOT; };
  auto swz  = [&](const float* s, int i) {
    swizzleB<<<g1(WSLOT), 256, 0, stream>>>(s, slot(i), DIM);
  };
  auto gemm = [&](const _Float16* Am, const _Float16* Bm,
                  const _Float16* A2m, const _Float16* B2m,
                  const float* bias, const float* pre, const float* post, float* o,
                  int M, int K, int K2, int act) {
    gemm_wmma<<<(M + 31) / 32, 256, 0, stream>>>(Am, Bm, A2m, B2m, bias, pre, post, o,
                                                 M, DIM, K, K2, act);
  };

  // ---- weight swizzles (f32 -> fragment-ordered f16), once per launch ----
  enum { S_W0 = 0, S_dW0, S_dW1, S_Wb, S_Wg0, S_Wg1, S_Ws0, S_Ws1, S_Wp0, S_Wp1,
         S_uW0a, S_uW0b, S_uW1a, S_uW1b, S_WeA, S_WeB };
  swz(W0, S_W0);
  swz(dW, S_dW0);               swz(dW + WSLOT, S_dW1);
  swz(Wb, S_Wb);
  swz(agWg, S_Wg0);             swz(agWg + WSLOT, S_Wg1);
  swz(agWs, S_Ws0);             swz(agWs + WSLOT, S_Ws1);
  swz(agWp, S_Wp0);             swz(agWp + WSLOT, S_Wp1);
  swz(uW, S_uW0a);              swz(uW + WSLOT, S_uW0b);
  swz(uW + 2 * WSLOT, S_uW1a);  swz(uW + 3 * WSLOT, S_uW1b);
  swz(We, S_WeA);               swz(We + WSLOT, S_WeB);

  // X = relu(X@W0+b0); start = X (second output)
  cvt(X_in, AhA, NODES * DIM);
  gemm(AhA, slot(S_W0), nullptr, nullptr, b0, nullptr, nullptr, start, NODES, DIM, 0, 1);

  // ---- level 0 (n = 4096, adjacency = A_in) ----
  cvt(start, AhA, NODES * DIM);
  gemm(AhA, slot(S_dW0), nullptr, nullptr, db, nullptr, nullptr, down0, NODES, DIM, 0, 1);
  matvec_bias2<<<g1(NODES), 256, 0, stream>>>(down0, pW, pwb, pb, 0, h, NODES, DIM);
  pool_scores<<<NODES, 256, 0, stream>>>(A_in, h, phi, 0, NODES, scores);
  topk_rank<<<g1(NODES), 256, 0, stream>>>(scores, NODES, NL1, idx0, vals0);
  gather_rows_scale<<<g1(NL1 * DIM), 256, 0, stream>>>(down0, idx0, vals0, G0, NL1, DIM);
  gather_adj<<<g1(NL1 * NL1), 256, 0, stream>>>(A_in, idx0, Ap1, NL1, NODES);

  // ---- level 1 (n = 3276, adjacency = Ap1) ----
  cvt(G0, AhA, NL1 * DIM);
  gemm(AhA, slot(S_dW1), nullptr, nullptr, db + DIM, nullptr, nullptr, down1, NL1, DIM, 0, 1);
  matvec_bias2<<<g1(NL1), 256, 0, stream>>>(down1, pW + DIM, pwb, pb, 1, h, NL1, DIM);
  pool_scores<<<NL1, 256, 0, stream>>>(Ap1, h, phi, 1, NL1, scores);
  topk_rank<<<g1(NL1), 256, 0, stream>>>(scores, NL1, NL2, idx1, vals1);
  gather_rows_scale<<<g1(NL2 * DIM), 256, 0, stream>>>(down1, idx1, vals1, G1, NL2, DIM);
  // (A[idx1][:,idx1] is never consumed by the reference -> skipped)

  // ---- bottleneck ----
  cvt(G1, AhA, NL2 * DIM);
  gemm(AhA, slot(S_Wb), nullptr, nullptr, bb, nullptr, nullptr, G2, NL2, DIM, 0, 1);

  // ---- up step 0 (u=1): n = 3276, idx1, down1 ----
  zero_buf<<<g1(NL1 * DIM), 256, 0, stream>>>(G3, NL1 * DIM);
  scatter_rows<<<g1(NL2 * DIM), 256, 0, stream>>>(G2, idx1, G3, NL2, DIM);      // Xf
  cvt(G3, AhA, NL1 * DIM);
  gemm(AhA, slot(S_Wg0), nullptr, nullptr, agbg, nullptr, nullptr, G0, NL1, DIM, 0, 0);  // T1
  cvt(down1, AhB, NL1 * DIM);
  gemm(AhB, slot(S_Ws0), nullptr, nullptr, agbs, G0, nullptr, G1, NL1, DIM, 0, 1);       // T2
  cvt(G1, AhB, NL1 * DIM);
  gemm(AhB, slot(S_Wp0), nullptr, nullptr, agbp, nullptr, nullptr, G2, NL1, DIM, 0, 2);  // gate
  cvt(G2, AhB, NL1 * DIM);   // AhA still holds Xf
  gemm(AhA, slot(S_uW0a), AhB, slot(S_uW0b), ub, nullptr, down1, G4, NL1, DIM, DIM, 1);  // Xu1

  // ---- up step 1 (u=0): n = 4096, idx0, down0 ----
  zero_buf<<<g1(NODES * DIM), 256, 0, stream>>>(G5, NODES * DIM);
  scatter_rows<<<g1(NL1 * DIM), 256, 0, stream>>>(G4, idx0, G5, NL1, DIM);      // Xf0
  cvt(G5, AhA, NODES * DIM);
  gemm(AhA, slot(S_Wg1), nullptr, nullptr, agbg + DIM, nullptr, nullptr, G0, NODES, DIM, 0, 0);
  cvt(down0, AhB, NODES * DIM);
  gemm(AhB, slot(S_Ws1), nullptr, nullptr, agbs + DIM, G0, nullptr, G1, NODES, DIM, 0, 1);
  cvt(G1, AhB, NODES * DIM);
  gemm(AhB, slot(S_Wp1), nullptr, nullptr, agbp + DIM, nullptr, nullptr, G2, NODES, DIM, 0, 2);
  cvt(G2, AhB, NODES * DIM);  // AhA still holds Xf0
  gemm(AhA, slot(S_uW1a), AhB, slot(S_uW1b), ub + DIM, nullptr, down0, G3, NODES, DIM, DIM, 1);

  // ---- head: out = relu(concat([Xu0, start]) @ We + be) ----
  cvt(G3, AhA, NODES * DIM);
  cvt(start, AhB, NODES * DIM);
  gemm(AhA, slot(S_WeA), AhB, slot(S_WeB), be, nullptr, nullptr, out, NODES, DIM, DIM, 1);
}
